// BayesMultiheadAttention_38062000177313
// MI455X (gfx1250) — matrix operand order repaired and checked by reading, not compiled, not run on hardware
//
#include <hip/hip_runtime.h>
#include <hip/hip_bf16.h>

// ---------------------------------------------------------------------------
// BayesMultiheadAttention for MI455X (gfx1250, wave32, WMMA bf16 path)
//   B=4, T=2048, D=1024, H=8, HD=128
// Pipeline:
//   k_qkv_proj   : double-buffered LDS block-tiled GEMM,
//                  Q/K/V = (x @ W^T) * head_mask -> bf16 (B*H, T, HD).
//                  Block tile 128(M) x 64(N), K-step 32; global prefetch of
//                  tile s+1 overlaps WMMA of tile s; 1 barrier / K-step.
//   k_flash_attn : causal online-softmax attention -> f32 (B*H, T, HD).
//                  K chunk staged via GLOBAL_LOAD_ASYNC_TO_LDS_B128 (ASYNCcnt),
//                  V chunk transposed through VGPRs; all WMMA operand fetches
//                  are ds_load_b128 pairs.
//   k_head_prod  : prod over heads -> bf16 (B*T, HD)
//   k_out_proj   : @ Wout^T -> f32 (B, T, D)
// ---------------------------------------------------------------------------

typedef __bf16 bf16_t;
typedef __attribute__((ext_vector_type(16))) __bf16 v16bf;
typedef __attribute__((ext_vector_type(8)))  float  v8f;

constexpr int B_ = 4, T_ = 2048, D_ = 1024, H_ = 8, HD_ = 128;
constexpr int BT_ = B_ * T_;

__device__ __forceinline__ v16bf pack2(uint4 lo, uint4 hi) {
  union { uint4 u[2]; v16bf v; } c;
  c.u[0] = lo; c.u[1] = hi;
  return c.v;
}

// A-matrix fragment (16x32, MxK), bf16 source, two 16B loads.
// ISA layout: lane m = lane&15; lanes<16 hold K {0..7,16..23}, lanes>=16 hold
// K {8..15,24..31}.
__device__ __forceinline__ v16bf a_frag_bf16(const bf16_t* __restrict__ base,
                                             int ld, int kbase, int lane) {
  const int m = lane & 15, g = lane >> 4;
  const bf16_t* p = base + (size_t)m * ld + kbase + g * 8;
  uint4 lo = *(const uint4*)(p);
  uint4 hi = *(const uint4*)(p + 16);
  return pack2(lo, hi);
}

// B-matrix fragment (32x16, KxN) for A @ W^T with W row-major (N x K):
// lane holds column n = lane&15; lanes<16 hold K {0..7,16..23}, lanes>=16 K {8..15,24..31}.
__device__ __forceinline__ v16bf bT_frag_bf16(const bf16_t* __restrict__ w,
                                              int ld, int kbase, int lane) {
  const int n = lane & 15, g = lane >> 4;
  const bf16_t* p = w + (size_t)n * ld + kbase + g * 8;
  uint4 lo = *(const uint4*)(p);
  uint4 hi = *(const uint4*)(p + 16);
  return pack2(lo, hi);
}

// f32 source variant (used only where K is tiny: out-proj B operand)
__device__ __forceinline__ v16bf bT_frag_f32(const float* __restrict__ w,
                                             int ld, int kbase, int lane) {
  const int n = lane & 15, g = lane >> 4;
  const float* p = w + (size_t)n * ld + kbase + g * 8;
  v16bf b;
#pragma unroll
  for (int j = 0; j < 16; ++j) b[j] = (bf16_t)p[j < 8 ? j : j + 8];
  return b;
}

__device__ __forceinline__ v8f wmma_bf16(v16bf a, v16bf b, v8f c) {
  return __builtin_amdgcn_wmma_f32_16x16x32_bf16(
      /*neg_a=*/false, a, /*neg_b=*/false, b,
      /*c_mod=*/(short)0, c, /*reuse_a=*/false, /*reuse_b=*/false);
}

// ---------------------------------------------------------------------------
// Kernel 1: QKV projection, double-buffered LDS block-tiled GEMM.
//   grid.x: (BT/128) * (D/64) block tiles; grid.z in {0,1,2} selects Q/K/V.
//   8 waves; wave w owns output rows [row0+16w, row0+16w+16) x all 64 cols.
// ---------------------------------------------------------------------------
__global__ __launch_bounds__(256) void k_qkv_proj(
    const float* __restrict__ x,
    const float* __restrict__ Wq, const float* __restrict__ Wk,
    const float* __restrict__ Wv,
    const float* __restrict__ mq, const float* __restrict__ mk,
    const float* __restrict__ mv,
    bf16_t* __restrict__ Qb, bf16_t* __restrict__ Kb, bf16_t* __restrict__ Vb)
{
  __shared__ __align__(16) bf16_t Xs[2][128][32];  // 16 KB ping-pong
  __shared__ __align__(16) bf16_t Ws[2][64][32];   //  8 KB ping-pong

  const int tid  = threadIdx.x;
  const int lane = tid & 31;
  const int wave = tid >> 5;
  const int sel  = blockIdx.z;
  const float* W  = (sel == 0) ? Wq : (sel == 1) ? Wk : Wv;
  const float* Mh = (sel == 0) ? mq : (sel == 1) ? mk : mv;
  bf16_t*      Ob = (sel == 0) ? Qb : (sel == 1) ? Kb : Vb;

  const int ntN  = D_ / 64;                        // 16 N block-tiles
  const int mt   = blockIdx.x / ntN;               // 0..63
  const int nt   = blockIdx.x % ntN;
  const int row0 = mt * 128, col0 = nt * 64;

  v8f acc[4];
#pragma unroll
  for (int c = 0; c < 4; ++c) acc[c] = (v8f){};

  // cooperative-load index precompute
  const int xr = tid >> 1;                 // 0..127 : row in x tile
  const int xc = (tid & 1) * 16;           // col half (16 f32 each)
  const int wr = tid >> 2;                 // 0..63  : row in W tile
  const int wc = (tid & 3) * 8;            // col quarter (8 f32 each)
  const float* xrow = x + (size_t)(row0 + xr) * D_;
  const float* wrow = W + (size_t)(col0 + wr) * D_;

  float xf[16], wf[8];
  // prologue: tile 0 -> regs -> LDS buffer 0
#pragma unroll
  for (int j = 0; j < 16; ++j) xf[j] = xrow[xc + j];
#pragma unroll
  for (int j = 0; j < 8;  ++j) wf[j] = wrow[wc + j];
#pragma unroll
  for (int j = 0; j < 16; ++j) Xs[0][xr][xc + j] = (bf16_t)xf[j];
#pragma unroll
  for (int j = 0; j < 8;  ++j) Ws[0][wr][wc + j] = (bf16_t)wf[j];
  __syncthreads();

  const int nsteps = D_ / 32;                      // 32 K-steps
  for (int s = 0; s < nsteps; ++s) {
    const int buf = s & 1;
    if (s + 1 < nsteps) {                          // prefetch tile s+1
      const int kb = (s + 1) * 32;
#pragma unroll
      for (int j = 0; j < 16; ++j) xf[j] = xrow[kb + xc + j];
#pragma unroll
      for (int j = 0; j < 8;  ++j) wf[j] = wrow[kb + wc + j];
    }
    // compute tile s out of LDS (overlaps the prefetch above)
    const v16bf a = a_frag_bf16(&Xs[buf][wave * 16][0], 32, 0, lane);
#pragma unroll
    for (int c = 0; c < 4; ++c)
      acc[c] = wmma_bf16(a, bT_frag_bf16(&Ws[buf][c * 16][0], 32, 0, lane),
                         acc[c]);
    if (s + 1 < nsteps) {                          // publish tile s+1
      const int nb = buf ^ 1;
#pragma unroll
      for (int j = 0; j < 16; ++j) Xs[nb][xr][xc + j] = (bf16_t)xf[j];
#pragma unroll
      for (int j = 0; j < 8;  ++j) Ws[nb][wr][wc + j] = (bf16_t)wf[j];
      __syncthreads();
    }
  }

  // epilogue: per-(b,head,hd) dropout mask, scatter to (B*H, T, HD) bf16
  const int n = lane & 15, g = lane >> 4;
  const int b = row0 >> 11;                        // / T (128 | 2048: same b)
  const int trow0 = row0 + wave * 16 - b * T_;
#pragma unroll
  for (int c = 0; c < 4; ++c) {
    const int dout = col0 + c * 16 + n;
    const int h  = dout >> 7;
    const int hd = dout & (HD_ - 1);
    const float mval = Mh[((size_t)b * H_ + h) * HD_ + hd];
#pragma unroll
    for (int r = 0; r < 8; ++r) {
      const int t = trow0 + g * 8 + r;
      Ob[(((size_t)b * H_ + h) * T_ + t) * HD_ + hd] = (bf16_t)(acc[c][r] * mval);
    }
  }
}

// ---------------------------------------------------------------------------
// Kernel 2: causal flash attention. One block per (b*H+h, 128-row q tile);
// each of the 8 waves owns 16 q rows. K chunk DMA'd straight into LDS with
// GLOBAL_LOAD_ASYNC_TO_LDS_B128 (no VGPR round-trip, ASYNCcnt-tracked),
// overlapping the VGPR transpose of the V chunk.
// ---------------------------------------------------------------------------
__global__ __launch_bounds__(256) void k_flash_attn(
    const bf16_t* __restrict__ Qb, const bf16_t* __restrict__ Kb,
    const bf16_t* __restrict__ Vb, float* __restrict__ O)
{
  __shared__ __align__(16) bf16_t Kl[32][HD_];     // 8 KB (key-major)
  __shared__ __align__(16) bf16_t Vt[HD_][32];     // 8 KB (hd-major: transposed)
  __shared__ __align__(16) bf16_t Pl[8][16][32];   // 8 KB, wave-private slabs

  const int lane = threadIdx.x & 31;
  const int wave = threadIdx.x >> 5;
  const int qt = blockIdx.x & 15;                  // T/128 = 16 q tiles
  const int bh = blockIdx.x >> 4;                  // b*H + h
  const int q0 = qt * 128;

  const bf16_t* Qp = Qb + ((size_t)bh * T_ + q0 + wave * 16) * HD_;
  const bf16_t* Kp = Kb + (size_t)bh * T_ * HD_;
  const bf16_t* Vp = Vb + (size_t)bh * T_ * HD_;
  const unsigned long long ksaddr = (unsigned long long)(uintptr_t)Kp;

  v16bf qf[4];
#pragma unroll
  for (int i = 0; i < 4; ++i) qf[i] = a_frag_bf16(Qp, HD_, 32 * i, lane);

  v8f acc[8];
#pragma unroll
  for (int t = 0; t < 8; ++t) acc[t] = (v8f){};
  float mrow[8], lrow[8];
#pragma unroll
  for (int r = 0; r < 8; ++r) { mrow[r] = -1.0e30f; lrow[r] = 0.0f; }

  const int g = lane >> 4, n = lane & 15;
  const int qrow_base = q0 + wave * 16 + g * 8;
  const float scl = 0.0883883476483184f;           // 1/sqrt(HD)

  // per-thread chunk-copy indices
  const int r32 = threadIdx.x >> 3;                // key row 0..31
  const int seg = (threadIdx.x & 7) * 16;          // hd segment

  const int nchunks = (q0 + 128) / 32;             // causal frontier
  for (int kc = 0; kc < nchunks; ++kc) {
    const int kb = kc * 32;
    __syncthreads();
    {
      // K: async DMA global -> LDS (two b128 per thread), ASYNCcnt-tracked
      const unsigned int voff =
          (unsigned int)(((kb + r32) * HD_ + seg) * sizeof(bf16_t));
      const unsigned int ldst = (unsigned int)(uintptr_t)&Kl[r32][seg];
      asm volatile("global_load_async_to_lds_b128 %0, %1, %2"
                   :: "v"(ldst), "v"(voff), "s"(ksaddr) : "memory");
      asm volatile("global_load_async_to_lds_b128 %0, %1, %2"
                   :: "v"(ldst + 16u), "v"(voff + 16u), "s"(ksaddr) : "memory");

      // V: through VGPRs with transpose (overlaps the async K DMA)
      union { uint4 u[2]; bf16_t e[16]; } vv;
      const uint4* vs = (const uint4*)(Vp + (size_t)(kb + r32) * HD_ + seg);
      vv.u[0] = vs[0]; vv.u[1] = vs[1];
#pragma unroll
      for (int j = 0; j < 16; ++j) Vt[seg + j][r32] = vv.e[j];

      asm volatile("s_wait_asynccnt 0x0" ::: "memory");
    }
    __syncthreads();

    // S = Q K^T over HD=128: two 16x16 score tiles, 4 WMMA each
    v8f s0 = {}, s1 = {};
#pragma unroll
    for (int i = 0; i < 4; ++i) {
      s0 = wmma_bf16(qf[i], bT_frag_bf16(&Kl[0][0],  HD_, 32 * i, lane), s0);
      s1 = wmma_bf16(qf[i], bT_frag_bf16(&Kl[16][0], HD_, 32 * i, lane), s1);
    }

    // causal mask + online softmax (row reductions across 16-lane groups)
    float alpha[8];
#pragma unroll
    for (int r = 0; r < 8; ++r) {
      const int qrow = qrow_base + r;
      float v0 = s0[r] * scl, v1 = s1[r] * scl;
      if (kb      + n > qrow) v0 = -1.0e30f;
      if (kb + 16 + n > qrow) v1 = -1.0e30f;
      float mx = fmaxf(v0, v1);
#pragma unroll
      for (int off = 1; off < 16; off <<= 1)
        mx = fmaxf(mx, __shfl_xor(mx, off, 16));
      const float mnew = fmaxf(mrow[r], mx);
      alpha[r] = __expf(mrow[r] - mnew);
      const float p0 = __expf(v0 - mnew);
      const float p1 = __expf(v1 - mnew);
      float rs = p0 + p1;
#pragma unroll
      for (int off = 1; off < 16; off <<= 1)
        rs += __shfl_xor(rs, off, 16);
      lrow[r] = lrow[r] * alpha[r] + rs;
      mrow[r] = mnew;
      s0[r] = p0; s1[r] = p1;
    }
#pragma unroll
    for (int t = 0; t < 8; ++t)
#pragma unroll
      for (int r = 0; r < 8; ++r) acc[t][r] *= alpha[r];

    // C-layout -> A-fragment transpose of P through wave-private LDS slab.
    // (LDS ops are in-order within a wave; no barrier needed.)
#pragma unroll
    for (int r = 0; r < 8; ++r) {
      Pl[wave][g * 8 + r][n]      = (bf16_t)s0[r];
      Pl[wave][g * 8 + r][16 + n] = (bf16_t)s1[r];
    }
    const v16bf pf = a_frag_bf16(&Pl[wave][0][0], 32, 0, lane);

    // O += P @ V : one WMMA per 16-wide HD tile, B-frags from transposed V
#pragma unroll
    for (int t = 0; t < 8; ++t)
      acc[t] = wmma_bf16(pf, bT_frag_bf16(&Vt[t * 16][0], 32, 0, lane), acc[t]);
  }

  // normalize and store f32 (B*H, T, HD)
  float* Op = O + ((size_t)bh * T_ + q0 + wave * 16) * HD_;
#pragma unroll
  for (int r = 0; r < 8; ++r) {
    const float inv = 1.0f / lrow[r];
    const int rowM = g * 8 + r;
#pragma unroll
    for (int t = 0; t < 8; ++t)
      Op[(size_t)rowM * HD_ + t * 16 + n] = acc[t][r] * inv;
  }
}

// ---------------------------------------------------------------------------
// Kernel 3: product over heads -> bf16 (B*T, HD)
// ---------------------------------------------------------------------------
__global__ __launch_bounds__(256) void k_head_prod(
    const float* __restrict__ O, bf16_t* __restrict__ P)
{
  const int idx = blockIdx.x * 256 + threadIdx.x;  // B*T*HD = 2^20
  const int hd = idx & (HD_ - 1);
  const int t  = (idx >> 7) & (T_ - 1);
  const int b  = idx >> 18;                        // / (T*HD)
  float p = 1.0f;
#pragma unroll
  for (int h = 0; h < H_; ++h)
    p *= O[(((size_t)b * H_ + h) * T_ + t) * HD_ + hd];
  P[idx] = (bf16_t)p;
}

// ---------------------------------------------------------------------------
// Kernel 4: output projection (BT x 128) @ Wout^T (128 x 1024) -> f32 out
// ---------------------------------------------------------------------------
__global__ __launch_bounds__(256) void k_out_proj(
    const bf16_t* __restrict__ P, const float* __restrict__ Wout,
    float* __restrict__ out)
{
  const int lane = threadIdx.x & 31;
  const int wave = threadIdx.x >> 5;
  const int ntilesN = D_ / 16;                     // 64
  const int tile = blockIdx.x * 8 + wave;
  const int mt = tile / ntilesN, nt = tile % ntilesN;
  const int row0 = mt * 16, col0 = nt * 16;

  const bf16_t* abase = P + (size_t)row0 * HD_;
  const float*  bbase = Wout + (size_t)col0 * HD_;

  v8f acc = {};
#pragma unroll
  for (int kb = 0; kb < HD_; kb += 32)
    acc = wmma_bf16(a_frag_bf16(abase, HD_, kb, lane),
                    bT_frag_f32(bbase, HD_, kb, lane), acc);

  const int n = lane & 15, g = lane >> 4;
#pragma unroll
  for (int r = 0; r < 8; ++r)
    out[(size_t)(row0 + g * 8 + r) * D_ + col0 + n] = acc[r];
}

// ---------------------------------------------------------------------------
extern "C" void kernel_launch(void* const* d_in, const int* in_sizes, int n_in,
                              void* d_out, int out_size, void* d_ws, size_t ws_size,
                              hipStream_t stream) {
  (void)in_sizes; (void)n_in; (void)out_size; (void)ws_size;
  const float* x    = (const float*)d_in[0];
  const float* Wq   = (const float*)d_in[1];
  const float* Wk   = (const float*)d_in[2];
  const float* Wv   = (const float*)d_in[3];
  const float* Wout = (const float*)d_in[4];
  const float* mq   = (const float*)d_in[5];
  const float* mk   = (const float*)d_in[6];
  const float* mv   = (const float*)d_in[7];
  // d_in[8]: boolean causal mask -- causality is computed analytically.

  // workspace: Q/K/V bf16 (16 MB each) + O f32 (32 MB) + prod bf16 (2 MB)
  const size_t nQ = (size_t)B_ * H_ * T_ * HD_;    // 8,388,608
  char* ws = (char*)d_ws;
  bf16_t* Qb = (bf16_t*)ws;  ws += nQ * sizeof(bf16_t);
  bf16_t* Kb = (bf16_t*)ws;  ws += nQ * sizeof(bf16_t);
  bf16_t* Vb = (bf16_t*)ws;  ws += nQ * sizeof(bf16_t);
  float*  O  = (float*) ws;  ws += nQ * sizeof(float);
  bf16_t* Pp = (bf16_t*)ws;

  const dim3 blk(256);
  k_qkv_proj <<<dim3((BT_ / 128) * (D_ / 64), 1, 3), blk, 0, stream>>>(
      x, Wq, Wk, Wv, mq, mk, mv, Qb, Kb, Vb);
  k_flash_attn<<<dim3(B_ * H_ * (T_ / 128)), blk, 0, stream>>>(Qb, Kb, Vb, O);
  k_head_prod <<<dim3((B_ * T_ * HD_) / 256), blk, 0, stream>>>(O, Pp);
  k_out_proj  <<<dim3((BT_ / 16) * (D_ / 16) / 8), blk, 0, stream>>>(
      Pp, Wout, (float*)d_out);
}
